// LU_M2SR_15504831938731
// MI455X (gfx1250) — compile-verified, matrix operand order checked by reading, not run on hardware
//
#include <hip/hip_runtime.h>
#include <math.h>

#define HH 256
#define WW 256
#define BB 8
#define CC 64
#define HWSZ 65536

typedef __attribute__((ext_vector_type(16))) _Float16 v16h;
typedef __attribute__((ext_vector_type(8)))  _Float16 h8v;
typedef __attribute__((ext_vector_type(8)))  float    v8f;

#if __has_builtin(__builtin_amdgcn_global_load_async_to_lds_b32)
#define USE_ASYNC_LDS 1
#endif

typedef __attribute__((address_space(1))) int* gptr_t;
typedef __attribute__((address_space(3))) int* lptr_t;

static __device__ __forceinline__ float silu_f(float v) {
    // v * sigmoid(v) with native v_rcp_f32 (no IEEE div expansion)
    return v * __builtin_amdgcn_rcpf(1.0f + __expf(-v));
}

// Load 16 f16 WMMA operand elements for one lane:
// elements 0..7  = K .. K+7   (VGPRs 0-3)
// elements 8..15 = K+16..K+23 (VGPRs 4-7)
static __device__ __forceinline__ v16h ld16(const _Float16* p) {
    h8v lo = *(const h8v*)p;
    h8v hi = *(const h8v*)(p + 16);
    return __builtin_shufflevector(lo, hi, 0,1,2,3,4,5,6,7,8,9,10,11,12,13,14,15);
}

static __device__ __forceinline__ void wait_async0() {
#if __has_builtin(__builtin_amdgcn_s_wait_asynccnt)
    __builtin_amdgcn_s_wait_asynccnt(0);
#else
    asm volatile("s_wait_asynccnt 0" ::: "memory");
#endif
}

// ---------------------------------------------------------------------------
// Fused: h0 = silu(conv1(x)); dwcat = [dw1,dw2,dw3](h0); h = silu(silu(pw(dwcat)+b)+x)
// One workgroup = one 16x16 output tile (+2 halo) of one image. 256 threads.
// LDS carve (dynamic): [0, 98304)      : x halo f16 [400][64]  /  dwcat f16 [256][192]
//                      [98304, +102400): x halo f32 staging [64][400], then h0 f32 [64][400]
//                      [200704, +24576): conv1 W f16 [64][64] then pw W f16 [64][192]
// ---------------------------------------------------------------------------
__global__ __launch_bounds__(256) void fused_main(
    const float* __restrict__ x,
    const float* __restrict__ c1w, const float* __restrict__ c1b,
    const float* __restrict__ d1w, const float* __restrict__ d1b,
    const float* __restrict__ d2w, const float* __restrict__ d2b,
    const float* __restrict__ d3w, const float* __restrict__ d3b,
    const float* __restrict__ pww, const float* __restrict__ pwb,
    float* __restrict__ hout)
{
    extern __shared__ char smem[];
    _Float16* r0 = (_Float16*)smem;                 // x tile [p][ch] then dwcat [p][192]
    float*    h0 = (float*)(smem + 98304);          // f32 stage, then h0 [ch][400]
    _Float16* wl = (_Float16*)(smem + 200704);      // weights

    const int tid  = threadIdx.x;
    const int ln   = tid & 15;
    const int hf   = (tid >> 4) & 1;                // lane/16 within wave32
    const int wv   = tid >> 5;                      // wave id 0..7
    const int b    = blockIdx.z;
    const int y0   = blockIdx.y * 16;
    const int x0   = blockIdx.x * 16;

    __builtin_prefetch(x + (size_t)(b * CC) * HWSZ + y0 * WW + x0, 0, 1);

    // conv1 weights -> f16 LDS, row-major [co][ci]
    for (int i = tid; i < 64 * 64; i += 256) wl[i] = (_Float16)c1w[i];

#ifdef USE_ASYNC_LDS
    // Stage x halo (20x20, 64 ch) straight into LDS (f32) with async-to-LDS DMA.
    for (int i = tid; i < 64 * 400; i += 256) {
        int ch = i / 400, p = i % 400;
        int gy = y0 - 2 + p / 20;
        int gx = x0 - 2 + p % 20;
        if ((unsigned)gy < HH && (unsigned)gx < WW) {
            __builtin_amdgcn_global_load_async_to_lds_b32(
                (gptr_t)(x + (size_t)(b * CC + ch) * HWSZ + gy * WW + gx),
                (lptr_t)(h0 + i),
                0, 0);
        } else {
            h0[i] = 0.0f;
        }
    }
    wait_async0();
    __syncthreads();
    // f32 [ch][400] -> GEMM-B f16 [p][ch]
    for (int i = tid; i < 64 * 400; i += 256) {
        int ch = i / 400, p = i % 400;
        r0[p * 64 + ch] = (_Float16)h0[i];
    }
#else
    // Fallback: load + convert through VGPRs
    for (int i = tid; i < 64 * 400; i += 256) {
        int ch = i / 400, p = i % 400;
        int gy = y0 - 2 + p / 20;
        int gx = x0 - 2 + p % 20;
        float v = 0.0f;
        if ((unsigned)gy < HH && (unsigned)gx < WW)
            v = x[(size_t)(b * CC + ch) * HWSZ + gy * WW + gx];
        r0[p * 64 + ch] = (_Float16)v;
    }
#endif
    __syncthreads();

    // ---- conv1 GEMM: M=64 (4 tiles) x N=400 (25 tiles), K=64 (2 steps) ----
    for (int t = wv; t < 100; t += 8) {
        int Mb = (t / 25) * 16;
        int Nb = (t % 25) * 16;
        v8f acc = {0.f, 0.f, 0.f, 0.f, 0.f, 0.f, 0.f, 0.f};
#pragma unroll
        for (int kk = 0; kk < 2; ++kk) {
            v16h a  = ld16(wl + (Mb + ln) * 64 + kk * 32 + hf * 8);
            v16h bm = ld16(r0 + (Nb + ln) * 64 + kk * 32 + hf * 8);
            acc = __builtin_amdgcn_wmma_f32_16x16x32_f16(
                false, a, false, bm, (short)0, acc, false, false);
        }
        int p = Nb + ln;
#pragma unroll
        for (int r = 0; r < 8; ++r) {
            int c = Mb + hf * 8 + r;
            h0[c * 400 + p] = silu_f(acc[r] + c1b[c]);
        }
    }
    __syncthreads();

    // pw weights -> f16 LDS (reuses conv1-W region)
    for (int i = tid; i < 64 * 192; i += 256) wl[i] = (_Float16)pww[i];

    // ---- depthwise 1x1 / 3x3 / 5x5 -> dwcat f16 [p][192] (reuses x region) ----
    {
        int c  = tid & 63;
        int pg = tid >> 6;
        float w1d = d1w[c], b1d = d1b[c];
        float w2d[9], w3d[25];
#pragma unroll
        for (int j = 0; j < 9; ++j)  w2d[j] = d2w[c * 9 + j];
        float b2d = d2b[c];
#pragma unroll
        for (int j = 0; j < 25; ++j) w3d[j] = d3w[c * 25 + j];
        float b3d = d3b[c];
        const float* hr = h0 + c * 400;
        for (int pp = pg; pp < 256; pp += 4) {
            int cy = (pp >> 4) + 2, cx = (pp & 15) + 2;
            float o1 = w1d * hr[cy * 20 + cx] + b1d;
            float o2 = b2d;
#pragma unroll
            for (int dy = 0; dy < 3; ++dy)
#pragma unroll
                for (int dx = 0; dx < 3; ++dx)
                    o2 += w2d[dy * 3 + dx] * hr[(cy - 1 + dy) * 20 + (cx - 1 + dx)];
            float o3 = b3d;
#pragma unroll
            for (int dy = 0; dy < 5; ++dy)
#pragma unroll
                for (int dx = 0; dx < 5; ++dx)
                    o3 += w3d[dy * 5 + dx] * hr[(cy - 2 + dy) * 20 + (cx - 2 + dx)];
            r0[pp * 192 +       c] = (_Float16)o1;
            r0[pp * 192 +  64 + c] = (_Float16)o2;
            r0[pp * 192 + 128 + c] = (_Float16)o3;
        }
    }
    __syncthreads();

    // ---- pw GEMM: M=64 (4 tiles) x N=256 (16 tiles), K=192 (6 steps) ----
    for (int t = wv; t < 64; t += 8) {
        int Mb = (t >> 4) * 16;
        int Nb = (t & 15) * 16;
        v8f acc = {0.f, 0.f, 0.f, 0.f, 0.f, 0.f, 0.f, 0.f};
#pragma unroll
        for (int kk = 0; kk < 6; ++kk) {
            v16h a  = ld16(wl + (Mb + ln) * 192 + kk * 32 + hf * 8);
            v16h bm = ld16(r0 + (Nb + ln) * 192 + kk * 32 + hf * 8);
            acc = __builtin_amdgcn_wmma_f32_16x16x32_f16(
                false, a, false, bm, (short)0, acc, false, false);
        }
        int p  = Nb + ln;
        int gy = y0 + (p >> 4), gx = x0 + (p & 15);
#pragma unroll
        for (int r = 0; r < 8; ++r) {
            int c = Mb + hf * 8 + r;
            size_t gi = (size_t)(b * CC + c) * HWSZ + gy * WW + gx;
            float v = silu_f(acc[r] + pwb[c]) + x[gi];
            hout[gi] = silu_f(v);
        }
    }
}

// ---------------- channel avg/max pool: one block per (b,c) ----------------
__global__ __launch_bounds__(256) void pool_kernel(
    const float* __restrict__ h, float* __restrict__ avgp, float* __restrict__ maxp)
{
    __shared__ float ss[256], sm[256];
    int c = blockIdx.x, b = blockIdx.y, tid = threadIdx.x;
    const float* p = h + (size_t)(b * CC + c) * HWSZ;
    float s = 0.f, m = -1e30f;
    for (int i = tid; i < HWSZ; i += 256) { float v = p[i]; s += v; m = fmaxf(m, v); }
    ss[tid] = s; sm[tid] = m; __syncthreads();
    for (int st = 128; st > 0; st >>= 1) {
        if (tid < st) { ss[tid] += ss[tid + st]; sm[tid] = fmaxf(sm[tid], sm[tid + st]); }
        __syncthreads();
    }
    if (tid == 0) { avgp[b * CC + c] = ss[0] / (float)HWSZ; maxp[b * CC + c] = sm[0]; }
}

// ---------------- channel attention MLP: ca = silu(W2 relu(W1 a) + W2 relu(W1 m)) ----
__global__ __launch_bounds__(256) void ca_kernel(
    const float* __restrict__ avgp, const float* __restrict__ maxp,
    const float* __restrict__ w1, const float* __restrict__ w2, float* __restrict__ ca)
{
    __shared__ float sva[512], svm[512], ha[64], hm[64];
    int tid = threadIdx.x;
    for (int i = tid; i < 512; i += 256) { sva[i] = avgp[i]; svm[i] = maxp[i]; }
    __syncthreads();
    if (tid < 64) {
        int b = tid >> 3, k = tid & 7;
        float s1 = 0.f, s2 = 0.f;
        for (int ci = 0; ci < 64; ++ci) {
            float w = w1[k * 64 + ci];
            s1 += w * sva[b * 64 + ci];
            s2 += w * svm[b * 64 + ci];
        }
        ha[tid] = fmaxf(s1, 0.f);
        hm[tid] = fmaxf(s2, 0.f);
    }
    __syncthreads();
    for (int i = tid; i < 512; i += 256) {
        int b = i >> 6, c = i & 63;
        float s = 0.f;
        for (int k = 0; k < 8; ++k) s += w2[c * 8 + k] * (ha[b * 8 + k] + hm[b * 8 + k]);
        ca[i] = silu_f(s);
    }
}

// ---------------- in-place out = ca*h; spatial mean/max over channels -------
__global__ __launch_bounds__(256) void ca_apply(
    float* __restrict__ h, const float* __restrict__ ca,
    float* __restrict__ spmean, float* __restrict__ spmax)
{
    int p = blockIdx.x * 256 + threadIdx.x;
    int b = p >> 16, pix = p & 65535;
    float s = 0.f, m = -1e30f;
    for (int c = 0; c < CC; ++c) {
        size_t gi = (size_t)(b * CC + c) * HWSZ + pix;
        float v = h[gi] * ca[b * CC + c];
        h[gi] = v; s += v; m = fmaxf(m, v);
    }
    spmean[p] = s * (1.0f / 64.0f);
    spmax[p]  = m;
}

// ---------------- spatial attention 7x7 + in-place multiply -----------------
__global__ __launch_bounds__(256) void sa_apply(
    float* __restrict__ h, const float* __restrict__ spmean,
    const float* __restrict__ spmax, const float* __restrict__ sw,
    const float* __restrict__ sb)
{
    int p = blockIdx.x * 256 + threadIdx.x;
    int b = p >> 16, pix = p & 65535;
    int y = pix >> 8, xx = pix & 255;
    float s = sb[0];
    for (int dy = 0; dy < 7; ++dy) {
        int ny = y - 3 + dy; if ((unsigned)ny >= HH) continue;
        for (int dx = 0; dx < 7; ++dx) {
            int nx = xx - 3 + dx; if ((unsigned)nx >= WW) continue;
            int q = (b << 16) + (ny << 8) + nx;
            s += sw[dy * 7 + dx] * spmean[q] + sw[49 + dy * 7 + dx] * spmax[q];
        }
    }
    float sa = silu_f(s);
    for (int c = 0; c < CC; ++c) {
        size_t gi = (size_t)(b * CC + c) * HWSZ + pix;
        h[gi] *= sa;
    }
}

// ---------------- GroupNorm stats: one block per (group,b) ------------------
__global__ __launch_bounds__(256) void gn_stats(
    const float* __restrict__ h, float* __restrict__ mu, float* __restrict__ rs)
{
    __shared__ float s1[256], s2[256];
    int g = blockIdx.x, b = blockIdx.y, tid = threadIdx.x;
    const float* p = h + (size_t)(b * CC + g * 8) * HWSZ;
    float a = 0.f, q = 0.f;
    for (int i = tid; i < 8 * HWSZ; i += 256) { float v = p[i]; a += v; q += v * v; }
    s1[tid] = a; s2[tid] = q; __syncthreads();
    for (int st = 128; st > 0; st >>= 1) {
        if (tid < st) { s1[tid] += s1[tid + st]; s2[tid] += s2[tid + st]; }
        __syncthreads();
    }
    if (tid == 0) {
        float n = (float)(8 * HWSZ);
        float m = s1[0] / n;
        float var = s2[0] / n - m * m;
        mu[b * 8 + g] = m;
        rs[b * 8 + g] = __builtin_amdgcn_rsqf(var + 1e-5f);
    }
}

// ---------------- GN normalize + SiLU, in place -----------------------------
__global__ __launch_bounds__(256) void gn_apply(
    float* __restrict__ h, const float* __restrict__ mu, const float* __restrict__ rs,
    const float* __restrict__ gg, const float* __restrict__ gb)
{
    size_t i = (size_t)blockIdx.x * 256 + threadIdx.x;
    int c = (int)((i >> 16) & 63);
    int b = (int)(i >> 22);
    int g = c >> 3;
    float v = (h[i] - mu[b * 8 + g]) * rs[b * 8 + g] * gg[c] + gb[c];
    h[i] = silu_f(v);
}

extern "C" void kernel_launch(void* const* d_in, const int* in_sizes, int n_in,
                              void* d_out, int out_size, void* d_ws, size_t ws_size,
                              hipStream_t stream) {
    const float* x    = (const float*)d_in[0];
    const float* c1w  = (const float*)d_in[1];
    const float* c1b  = (const float*)d_in[2];
    const float* d1w  = (const float*)d_in[3];
    const float* d1b  = (const float*)d_in[4];
    const float* d2w  = (const float*)d_in[5];
    const float* d2b  = (const float*)d_in[6];
    const float* d3w  = (const float*)d_in[7];
    const float* d3b  = (const float*)d_in[8];
    const float* pww  = (const float*)d_in[9];
    const float* pwb  = (const float*)d_in[10];
    const float* caw1 = (const float*)d_in[11];
    const float* caw2 = (const float*)d_in[12];
    const float* saw  = (const float*)d_in[13];
    const float* sab  = (const float*)d_in[14];
    const float* gng  = (const float*)d_in[15];
    const float* gnb  = (const float*)d_in[16];
    float* out = (float*)d_out;

    char* ws = (char*)d_ws;
    float* spmean = (float*)ws;                         // 2 MB
    float* spmax  = (float*)(ws + 2097152);             // 2 MB
    float* avgp   = (float*)(ws + 4194304);
    float* maxp   = (float*)(ws + 4194304 + 4096);
    float* cav    = (float*)(ws + 4194304 + 8192);
    float* gnmu   = (float*)(ws + 4194304 + 12288);
    float* gnrs   = (float*)(ws + 4194304 + 16384);

    (void)in_sizes; (void)n_in; (void)out_size; (void)ws_size;

    (void)hipFuncSetAttribute(reinterpret_cast<const void*>(fused_main),
                              hipFuncAttributeMaxDynamicSharedMemorySize, 225280);

    fused_main<<<dim3(16, 16, 8), 256, 225280, stream>>>(
        x, c1w, c1b, d1w, d1b, d2w, d2b, d3w, d3b, pww, pwb, out);
    pool_kernel<<<dim3(64, 8), 256, 0, stream>>>(out, avgp, maxp);
    ca_kernel<<<1, 256, 0, stream>>>(avgp, maxp, caw1, caw2, cav);
    ca_apply<<<2048, 256, 0, stream>>>(out, cav, spmean, spmax);
    sa_apply<<<2048, 256, 0, stream>>>(out, spmean, spmax, saw, sab);
    gn_stats<<<dim3(8, 8), 256, 0, stream>>>(out, gnmu, gnrs);
    gn_apply<<<131072, 256, 0, stream>>>(out, gnmu, gnrs, gng, gnb);
}
